// MultiSensorObsEmbedder_49435073577258
// MI455X (gfx1250) — compile-verified, matrix operand order checked by reading, not run on hardware
//
#include <hip/hip_runtime.h>

#define S_       4
#define META_    28
#define EDIM_    4
#define MLP_IN_  33
#define HID_     64
#define MLP_OUT_ 31
#define SED_     32
#define NPIX_    1024
#define NBK_     16
#define BT_      16
#define K_       512      // NBK*SED
#define N_       256      // OUT
#define M_       16384    // BT*NPIX
#define NSEG_    262144   // BT*NPIX*NBK

typedef __attribute__((ext_vector_type(16))) _Float16 v16h;
typedef __attribute__((ext_vector_type(8)))  _Float16 v8h;
typedef __attribute__((ext_vector_type(8)))  float    v8f;

// ---------------------------------------------------------------------------
// Kernel 1: transpose + fp16-convert agg_W : [S][K][N] f32 -> awT [S][N][K] f16
// ---------------------------------------------------------------------------
__global__ __launch_bounds__(256) void make_awT(
    const float* __restrict__ aggW, _Float16* __restrict__ awT)
{
    const int idx = blockIdx.x * 256 + threadIdx.x;
    if (idx >= S_ * K_ * N_) return;
    const int s   = idx / (K_ * N_);
    const int rem = idx - s * (K_ * N_);
    const int k   = rem / N_;
    const int n   = rem - k * N_;
    awT[((size_t)(s * N_ + n)) * K_ + k] = (_Float16)aggW[idx];
}

// ---------------------------------------------------------------------------
// Kernel 2: per-obs tokenizer MLP (33->64, LN, SiLU, 64->31) + scatter-add
// ---------------------------------------------------------------------------
__global__ __launch_bounds__(256) void obs_tokenize_scatter(
    const float* __restrict__ obs,
    const float* __restrict__ meta,
    const int*   __restrict__ pix,
    const int*   __restrict__ lch,
    const int*   __restrict__ lpl,
    const int*   __restrict__ otype,
    const int*   __restrict__ offsets,
    const float* __restrict__ etab,
    const float* __restrict__ W1,
    const float* __restrict__ b1,
    const float* __restrict__ lng,
    const float* __restrict__ lnb,
    const float* __restrict__ W2,
    const float* __restrict__ b2,
    float* __restrict__ sums,
    float* __restrict__ counts,
    int s, int Ns)
{
    __shared__ float sW1[MLP_IN_ * HID_];
    __shared__ float sW2[HID_ * MLP_OUT_];
    __shared__ float sb1[HID_], sg[HID_], sbt[HID_];
    __shared__ float sb2[MLP_OUT_];
    __shared__ int   sends[BT_];

    const int t = threadIdx.x;
    for (int i = t; i < MLP_IN_ * HID_; i += 256) sW1[i] = W1[s * MLP_IN_ * HID_ + i];
    for (int i = t; i < HID_ * MLP_OUT_; i += 256) sW2[i] = W2[s * HID_ * MLP_OUT_ + i];
    if (t < HID_) { sb1[t] = b1[s*HID_+t]; sg[t] = lng[s*HID_+t]; sbt[t] = lnb[s*HID_+t]; }
    if (t < MLP_OUT_) sb2[t] = b2[s*MLP_OUT_ + t];
    if (t < BT_) {
        const int start = (s == 0) ? 0 : offsets[s*BT_ - 1];
        sends[t] = offsets[s*BT_ + t] - start;   // local exclusive window ends
    }
    __syncthreads();

    const int i = blockIdx.x * 256 + t;
    if (i >= Ns) return;
    const long g = (long)s * Ns + i;

    const float o = obs[g];
    float x[MLP_IN_];
    x[0] = o;
    const float* mp = meta + g * META_;
    #pragma unroll
    for (int k = 0; k < META_; k++) x[1 + k] = mp[k];
    const float* ep = etab + ((long)s * 1024 + otype[g]) * EDIM_;
    #pragma unroll
    for (int k = 0; k < EDIM_; k++) x[1 + META_ + k] = ep[k];

    // h = x @ W1 + b1
    float h[HID_];
    #pragma unroll
    for (int j = 0; j < HID_; j++) h[j] = sb1[j];
    for (int k = 0; k < MLP_IN_; k++) {
        const float xv = x[k];
        const float* wr = &sW1[k * HID_];
        #pragma unroll
        for (int j = 0; j < HID_; j++) h[j] = fmaf(xv, wr[j], h[j]);
    }
    // LayerNorm + SiLU
    float mu = 0.f;
    #pragma unroll
    for (int j = 0; j < HID_; j++) mu += h[j];
    mu *= (1.0f / HID_);
    float var = 0.f;
    #pragma unroll
    for (int j = 0; j < HID_; j++) { const float d = h[j] - mu; var = fmaf(d, d, var); }
    var *= (1.0f / HID_);
    const float inv = rsqrtf(var + 1e-5f);
    #pragma unroll
    for (int j = 0; j < HID_; j++) {
        const float v = (h[j] - mu) * inv * sg[j] + sbt[j];
        h[j] = v / (1.0f + expf(-v));            // SiLU
    }
    // tok = [o, h @ W2 + b2]
    float tk[SED_];
    tk[0] = o;
    #pragma unroll
    for (int j2 = 0; j2 < MLP_OUT_; j2++) tk[1 + j2] = sb2[j2];
    for (int j = 0; j < HID_; j++) {
        const float hv = h[j];
        const float* wr = &sW2[j * MLP_OUT_];
        #pragma unroll
        for (int j2 = 0; j2 < MLP_OUT_; j2++) tk[1 + j2] = fmaf(hv, wr[j2], tk[1 + j2]);
    }
    // searchsorted(ends, i, 'right')
    int bidx = 0;
    #pragma unroll
    for (int q = 0; q < BT_; q++) bidx += (sends[q] <= i);
    const int bucket = lpl[g] * 8 + lch[g];
    const int seg = (bidx * NPIX_ + pix[g]) * NBK_ + bucket;

    float* sp = sums + (size_t)seg * SED_;
    #pragma unroll
    for (int d = 0; d < SED_; d++) unsafeAtomicAdd(sp + d, tk[d]);
    unsafeAtomicAdd(counts + seg, 1.0f);
}

// ---------------------------------------------------------------------------
// Kernel 3: out += 0.25 * ( mean-normalized feat (16384x512) @ aw (512x256) + b )
// WMMA f32_16x16x32_f16; block = 4 waves = 16 rows x 256 cols, 4 tiles/wave
// ---------------------------------------------------------------------------
__device__ inline v16h ldB16(const _Float16* p) {
    const v8h lo = *(const v8h*)p;
    const v8h hi = *(const v8h*)(p + 16);
    return __builtin_shufflevector(lo, hi, 0,1,2,3,4,5,6,7,8,9,10,11,12,13,14,15);
}

__global__ __launch_bounds__(128) void agg_gemm_wmma(
    const float*    __restrict__ sums,    // [M_][K_] segment sums
    const float*    __restrict__ counts,  // [M_*NBK_]
    const _Float16* __restrict__ awT,     // [S][N_][K_] f16
    const float*    __restrict__ aggb,    // [S][N_]
    float*          __restrict__ out,     // [M_][N_]
    int s)
{
    __shared__ _Float16 At[16 * K_];      // 16 KB normalized A tile (f16)

    const int t  = threadIdx.x;
    const int r0 = blockIdx.x * 16;

    // Stage 16 rows; divide each 32-wide chunk by its bucket count; cvt f16.
    for (int idx = t; idx < 16 * K_ / 4; idx += 128) {
        const int row = idx >> 7;              // /128 float4 per row
        const int c4  = idx & 127;
        const float4 v = ((const float4*)(sums + (size_t)(r0 + row) * K_))[c4];
        const float cnt = counts[(r0 + row) * NBK_ + (c4 >> 3)];   // bucket = (c4*4)>>5
        const float ic  = 1.0f / fmaxf(cnt, 1.0f);
        _Float16* dst = &At[row * K_ + c4 * 4];
        dst[0] = (_Float16)(v.x * ic); dst[1] = (_Float16)(v.y * ic);
        dst[2] = (_Float16)(v.z * ic); dst[3] = (_Float16)(v.w * ic);
    }
    __syncthreads();

    const int wv   = t >> 5;
    const int lane = t & 31;
    const int hl   = lane >> 4;     // lane half selects K sub-range
    const int l    = lane & 15;
    const int nb   = wv * 64;       // 4 N-tiles per wave

    v8f acc0 = {}, acc1 = {}, acc2 = {}, acc3 = {};
    const _Float16* bbase = awT + (size_t)s * N_ * K_;

    for (int k0 = 0; k0 < K_; k0 += 32) {
        const int ko = k0 + hl * 8;
        // A fragment: lane<16 -> K=k0+0..7,16..23 ; lane>=16 -> K=k0+8..15,24..31
        const v8h alo = *(const v8h*)&At[l * K_ + ko];
        const v8h ahi = *(const v8h*)&At[l * K_ + ko + 16];
        const v16h a = __builtin_shufflevector(alo, ahi,
                         0,1,2,3,4,5,6,7,8,9,10,11,12,13,14,15);

        const _Float16* bp = bbase + (size_t)(nb + l) * K_ + ko;
        __builtin_prefetch(bp + 32, 0, 1);     // next k-step of B
        acc0 = __builtin_amdgcn_wmma_f32_16x16x32_f16(
                   false, a, false, ldB16(bp),            (short)0, acc0, false, false);
        acc1 = __builtin_amdgcn_wmma_f32_16x16x32_f16(
                   false, a, false, ldB16(bp + 16 * K_),  (short)0, acc1, false, false);
        acc2 = __builtin_amdgcn_wmma_f32_16x16x32_f16(
                   false, a, false, ldB16(bp + 32 * K_),  (short)0, acc2, false, false);
        acc3 = __builtin_amdgcn_wmma_f32_16x16x32_f16(
                   false, a, false, ldB16(bp + 48 * K_),  (short)0, acc3, false, false);
    }

    // D layout: VGPR i -> rows (i, i+8) split by lane half; N = lane&15
    #pragma unroll
    for (int tt = 0; tt < 4; ++tt) {
        const v8f acc = (tt == 0) ? acc0 : (tt == 1) ? acc1 : (tt == 2) ? acc2 : acc3;
        const int col = nb + tt * 16 + l;
        const float bias = aggb[s * N_ + col];
        #pragma unroll
        for (int i2 = 0; i2 < 8; i2++) {
            const int r = r0 + hl * 8 + i2;
            const size_t oi = (size_t)r * N_ + col;
            out[oi] += 0.25f * (acc[i2] + bias);   // stream-ordered across sensors
        }
    }
}

// ---------------------------------------------------------------------------
extern "C" void kernel_launch(void* const* d_in, const int* in_sizes, int n_in,
                              void* d_out, int out_size, void* d_ws, size_t ws_size,
                              hipStream_t stream)
{
    (void)n_in; (void)ws_size;
    const float* obs     = (const float*)d_in[0];
    const float* meta    = (const float*)d_in[1];
    const int*   pix     = (const int*)d_in[2];
    const int*   lch     = (const int*)d_in[3];
    const int*   lpl     = (const int*)d_in[4];
    const int*   otype   = (const int*)d_in[5];
    const int*   offsets = (const int*)d_in[6];
    const float* etab    = (const float*)d_in[7];
    const float* W1      = (const float*)d_in[8];
    const float* b1      = (const float*)d_in[9];
    const float* lng     = (const float*)d_in[10];
    const float* lnb     = (const float*)d_in[11];
    const float* W2      = (const float*)d_in[12];
    const float* b2      = (const float*)d_in[13];
    const float* aggW    = (const float*)d_in[14];
    const float* aggb    = (const float*)d_in[15];

    const int nobs = in_sizes[0];
    const int Ns   = nobs / S_;

    // workspace layout: awT (1 MB) | sums (32 MB) | counts (1 MB)
    char* ws = (char*)d_ws;
    _Float16* awT = (_Float16*)ws;
    float* sums   = (float*)(ws + (1u << 20));
    float* counts = (float*)(ws + (1u << 20) + (size_t)NSEG_ * SED_ * sizeof(float));

    hipMemsetAsync(d_out, 0, (size_t)out_size * sizeof(float), stream);
    make_awT<<<(S_ * K_ * N_ + 255) / 256, 256, 0, stream>>>(aggW, awT);

    for (int s = 0; s < S_; s++) {
        // zero sums + counts (contiguous) for this sensor
        hipMemsetAsync(sums, 0, (size_t)NSEG_ * (SED_ + 1) * sizeof(float), stream);
        obs_tokenize_scatter<<<(Ns + 255) / 256, 256, 0, stream>>>(
            obs, meta, pix, lch, lpl, otype, offsets, etab,
            W1, b1, lng, lnb, W2, b2, sums, counts, s, Ns);
        agg_gemm_wmma<<<M_ / 16, 128, 0, stream>>>(
            sums, counts, awT, aggb, (float*)d_out, s);
    }
}
